// Network_31001073942734
// MI455X (gfx1250) — matrix operand order repaired and checked by reading, not compiled, not run on hardware
//
#include <hip/hip_runtime.h>
#include <hip/hip_bf16.h>

// ---------------------------------------------------------------------------
// Types for CDNA5 WMMA
// ---------------------------------------------------------------------------
typedef __attribute__((ext_vector_type(16))) __bf16 v16bf;
typedef __attribute__((ext_vector_type(8)))  float  v8f;

union V16 {
    v16bf        bf;
    unsigned int u[8];
    uint4        q[2];
};

__device__ __forceinline__ unsigned int bf16bits(float f) {
    unsigned int u = __builtin_bit_cast(unsigned int, f);
    // round-to-nearest-even truncation to bf16
    u = (u + 0x7FFFu + ((u >> 16) & 1u)) >> 16;
    return u & 0xFFFFu;
}
__device__ __forceinline__ unsigned int pack_bf2(float lo, float hi) {
    return bf16bits(lo) | (bf16bits(hi) << 16);
}

// guaranteed-hardware f32 atomic add (L2 atomic unit), no CAS fallback
__device__ __forceinline__ void atomic_add_f32(float* p, float v) {
    asm volatile("global_atomic_add_f32 %0, %1, off"
                 :
                 : "v"(p), "v"(v)
                 : "memory");
}

// ---------------------------------------------------------------------------
// z[R,64] = x[R,64] @ w[64,64]   (bf16 WMMA, f32 accumulate)
// block = 256 threads = 8 waves; waves laid out 2 row-tiles x 4 col-tiles
// block covers 32 rows x 64 cols of z.  Both operands staged in LDS as bf16.
// Row padding 72 (144B stride): 16B-aligned ds_load_b128, conflict-free
// across 16 consecutive rows (36 dwords/row; 9 coprime to 16).
// ---------------------------------------------------------------------------
__global__ void gemm64_wmma(const float* __restrict__ x,
                            const float* __restrict__ w,
                            float* __restrict__ z, int R) {
    __shared__ __align__(16) unsigned short wT[64][72]; // wT[n][k]
    __shared__ __align__(16) unsigned short xb[32][72]; // xb[row][k]
    const int tid    = threadIdx.x;
    const int m0base = blockIdx.x * 32;

    // ---- stage w transposed (64x64) as bf16
    for (int idx = tid; idx < 64 * 64; idx += 256) {
        int k = idx >> 6, n = idx & 63;
        wT[n][k] = (unsigned short)bf16bits(w[idx]);
    }
    // ---- stage x tile (32 rows x 64) as bf16: 8 elems per thread
    {
        int row  = tid >> 3;            // 0..31
        int col0 = (tid & 7) * 8;       // 0,8,..,56
        int gr   = m0base + row;
        int grc  = (gr < R) ? gr : (R - 1);
        const float* xr = x + (size_t)grc * 64 + col0;
        float4 a = *(const float4*)(xr);
        float4 b = *(const float4*)(xr + 4);
        uint4  pk;
        pk.x = pack_bf2(a.x, a.y); pk.y = pack_bf2(a.z, a.w);
        pk.z = pack_bf2(b.x, b.y); pk.w = pack_bf2(b.z, b.w);
        *(uint4*)&xb[row][col0] = pk;
    }
    __syncthreads();

    const int wave    = tid >> 5;
    const int lane    = tid & 31;
    const int rowTile = wave >> 2;       // 0..1
    const int colTile = wave & 3;        // 0..3
    const int m0      = m0base + rowTile * 16;
    const int n0      = colTile * 16;
    const int l15     = lane & 15;
    const int g       = lane >> 4;
    const int rloc    = rowTile * 16 + l15;   // local A row for this lane

    v8f acc = {};
#pragma unroll
    for (int c = 0; c < 2; ++c) {
        // ---- A operand: row rloc, K = 32c + {8g..8g+7, 16+8g..16+8g+7}
        V16 A;
        A.q[0] = *(const uint4*)&xb[rloc][32 * c + 8 * g];
        A.q[1] = *(const uint4*)&xb[rloc][32 * c + 16 + 8 * g];
        // ---- B operand: col n0+l15, K = 32c + {8g..8g+7, 16+8g..16+8g+7}
        V16 B;
        B.q[0] = *(const uint4*)&wT[n0 + l15][32 * c + 8 * g];
        B.q[1] = *(const uint4*)&wT[n0 + l15][32 * c + 16 + 8 * g];

        acc = __builtin_amdgcn_wmma_f32_16x16x32_bf16(
            false, A.bf, false, B.bf, (short)0, acc, false, false);
    }

    // D layout: lane l -> col n0+l15, VGPR j -> row m0 + j + 8*g
    float* zp = z + (size_t)(m0 + 8 * g) * 64 + n0 + l15;
    if (m0 + 16 <= R) {          // wave-uniform fast path: full tile
#pragma unroll
        for (int j = 0; j < 8; ++j) zp[(size_t)j * 64] = acc[j];
    } else {
#pragma unroll
        for (int j = 0; j < 8; ++j) {
            int mm = m0 + j + 8 * g;
            if (mm < R) z[(size_t)mm * 64 + n0 + l15] = acc[j];
        }
    }
}

// ---------------------------------------------------------------------------
// Layer-0 GEMM: tiny cin (<=5).  z[R,64] = x[R,cin] @ w[cin,64]
// ---------------------------------------------------------------------------
__global__ void small_gemm(const float* __restrict__ x,
                           const float* __restrict__ w,
                           float* __restrict__ z, int R, int cin) {
    long long idx = (long long)blockIdx.x * blockDim.x + threadIdx.x;
    if (idx >= (long long)R * 64) return;
    int r = (int)(idx >> 6), j = (int)(idx & 63);
    float a = 0.f;
    for (int k = 0; k < cin; ++k) a += x[(size_t)r * cin + k] * w[k * 64 + j];
    z[idx] = a;
}

// ---------------------------------------------------------------------------
// Edge scatter: acc[row] += val * z[col]  (one wave per edge, 2 ch per lane)
// ---------------------------------------------------------------------------
__global__ void scatter_add(const int* __restrict__ rows,
                            const int* __restrict__ cols,
                            const float* __restrict__ vals,
                            const float* __restrict__ z,
                            float* __restrict__ acc, int nnz) {
    long long t = (long long)blockIdx.x * blockDim.x + threadIdx.x;
    int e = (int)(t >> 5);
    if (e >= nnz) return;
    int lane = (int)(t & 31);
    int row = rows[e], col = cols[e];
    float  v  = vals[e];
    float2 zz = *(const float2*)(z + (size_t)col * 64 + lane * 2);
    float* a  = acc + (size_t)row * 64 + lane * 2;
    atomic_add_f32(a,     v * zz.x);
    atomic_add_f32(a + 1, v * zz.y);
}

// ---------------------------------------------------------------------------
// acc = tanh(acc) (+ cur if residual)    (in place)
// ---------------------------------------------------------------------------
__global__ void act_kernel(float* __restrict__ acc, const float* __restrict__ cur,
                           long long n, int residual) {
    long long i = (long long)blockIdx.x * blockDim.x + threadIdx.x;
    if (i >= n) return;
    float v = tanhf(acc[i]);
    if (residual) v += cur[i];
    acc[i] = v;
}

// ---------------------------------------------------------------------------
// Pool rank-0 features: per-channel sum / max / min  (64 blocks, 1 per ch)
// pool[c] = sum, pool[64+c] = max, pool[128+c] = min
// ---------------------------------------------------------------------------
__global__ void pool_kernel(const float* __restrict__ x0, int R,
                            float* __restrict__ pool) {
    const int c = blockIdx.x, tid = threadIdx.x;
    float s = 0.f, mx = -3.4e38f, mn = 3.4e38f;
    for (int r = tid; r < R; r += blockDim.x) {
        float v = x0[(size_t)r * 64 + c];
        s += v; mx = fmaxf(mx, v); mn = fminf(mn, v);
    }
    __shared__ float rs[256], rmx[256], rmn[256];
    rs[tid] = s; rmx[tid] = mx; rmn[tid] = mn;
    __syncthreads();
    for (int k = 128; k > 0; k >>= 1) {
        if (tid < k) {
            rs[tid] += rs[tid + k];
            rmx[tid] = fmaxf(rmx[tid], rmx[tid + k]);
            rmn[tid] = fminf(rmn[tid], rmn[tid + k]);
        }
        __syncthreads();
    }
    if (tid == 0) { pool[c] = rs[0]; pool[64 + c] = rmx[0]; pool[128 + c] = rmn[0]; }
}

// ---------------------------------------------------------------------------
// MLP head, single 512-thread block.
// ---------------------------------------------------------------------------
__device__ __forceinline__ void ln_leaky(float x, int K,
                                         const float* __restrict__ g,
                                         const float* __restrict__ b,
                                         float* __restrict__ h,
                                         float* __restrict__ red, int tid) {
    red[tid] = (tid < K) ? x : 0.f;
    __syncthreads();
    for (int s = 256; s > 0; s >>= 1) {
        if (tid < s) red[tid] += red[tid + s];
        __syncthreads();
    }
    float m = red[0] / (float)K;
    __syncthreads();
    float d = (tid < K) ? (x - m) : 0.f;
    red[tid] = d * d;
    __syncthreads();
    for (int s = 256; s > 0; s >>= 1) {
        if (tid < s) red[tid] += red[tid + s];
        __syncthreads();
    }
    float var = red[0] / (float)K;
    __syncthreads();
    if (tid < K) {
        float y = d * rsqrtf(var + 1e-5f) * g[tid] + b[tid];
        h[tid] = (y > 0.f) ? y : 0.2f * y;
    }
    __syncthreads();
}

__global__ void head_kernel(const float* __restrict__ pool,
                            const float* __restrict__ gf,
                            const float* __restrict__ W1, const float* __restrict__ B1,
                            const float* __restrict__ G1, const float* __restrict__ Bb1,
                            const float* __restrict__ W2, const float* __restrict__ B2,
                            const float* __restrict__ G2, const float* __restrict__ Bb2,
                            const float* __restrict__ W3, const float* __restrict__ B3,
                            const float* __restrict__ G3, const float* __restrict__ Bb3,
                            const float* __restrict__ W4, const float* __restrict__ B4,
                            float* __restrict__ out, float invN) {
    __shared__ float vec[260];
    __shared__ float h[512];
    __shared__ float red[512];
    const int tid = threadIdx.x;
    if (tid < 64) {
        float s = pool[tid];
        vec[tid]       = s * invN;        // mean
        vec[64 + tid]  = s;               // sum
        vec[128 + tid] = pool[64 + tid];  // max
        vec[192 + tid] = pool[128 + tid]; // min
    }
    if (tid < 4) vec[256 + tid] = gf[tid];
    __syncthreads();

    // fc1: 260 -> 512
    float v = 0.f;
    if (tid < 512) {
        v = B1[tid];
        for (int i = 0; i < 260; ++i) v += vec[i] * W1[i * 512 + tid];
    }
    ln_leaky(v, 512, G1, Bb1, h, red, tid);

    // fc2: 512 -> 256
    v = 0.f;
    if (tid < 256) {
        v = B2[tid];
        for (int i = 0; i < 512; ++i) v += h[i] * W2[i * 256 + tid];
    }
    ln_leaky(v, 256, G2, Bb2, h, red, tid);

    // fc3: 256 -> 128
    v = 0.f;
    if (tid < 128) {
        v = B3[tid];
        for (int i = 0; i < 256; ++i) v += h[i] * W3[i * 128 + tid];
    }
    ln_leaky(v, 128, G3, Bb3, h, red, tid);

    // fc4: 128 -> 2
    if (tid < 2) {
        float o = B4[tid];
        for (int i = 0; i < 128; ++i) o += h[i] * W4[i * 2 + tid];
        out[tid] = o;
    }
}

// ---------------------------------------------------------------------------
// Host-side orchestration
// ---------------------------------------------------------------------------
extern "C" void kernel_launch(void* const* d_in, const int* in_sizes, int n_in,
                              void* d_out, int out_size, void* d_ws, size_t ws_size,
                              hipStream_t stream) {
    static const int NS_[5]   = {50000, 100000, 200000, 50000, 10000};
    static const int INCH[5]  = {5, 1, 1, 1, 1};
    static const int ROFF[5]  = {0, 50000, 150000, 350000, 400000};
    static const int PS[15]   = {0, 0, 1, 0, 1, 2, 0, 1, 2, 3, 0, 1, 2, 3, 4};
    static const int PT[15]   = {0, 1, 1, 2, 2, 2, 3, 3, 3, 4, 4, 4, 4, 4};

    const size_t FEAT  = 410000ull * 64ull;   // total hidden elems, all ranks
    const size_t ZELEM = 200000ull * 64ull;   // largest z scratch
    if (ws_size < (2 * FEAT + ZELEM + 1024) * sizeof(float)) return;

    float* buf0 = (float*)d_ws;
    float* buf1 = buf0 + FEAT;
    float* zbuf = buf1 + FEAT;
    float* pool = zbuf + ZELEM;

    // input map (setup_inputs insertion order)
    const float* xin[5];
    for (int r = 0; r < 5; ++r) xin[r] = (const float*)d_in[r];
    const int*   rows[15]; const int* cols[15]; const float* vals[15];
    for (int p = 0; p < 15; ++p) {
        rows[p] = (const int*)d_in[5 + 2 * p];
        cols[p] = (const int*)d_in[6 + 2 * p];
        vals[p] = (const float*)d_in[35 + p];
    }
    const float* gf = (const float*)d_in[50];

    float* cur = nullptr;
    for (int li = 0; li < 3; ++li) {
        float* acc = (li == 1) ? buf1 : buf0;  // li0:buf0, li1:buf1, li2:buf0
        hipMemsetAsync(acc, 0, FEAT * sizeof(float), stream);
        for (int p = 0; p < 15; ++p) {
            const int s = PS[p], t = PT[p];
            const float* w  = (const float*)d_in[51 + 15 * li + p];
            const float* xs = (li == 0) ? xin[s] : cur + (size_t)ROFF[s] * 64;
            const int R = NS_[s];
            if (li == 0) {
                long long nthr = (long long)R * 64;
                small_gemm<<<(unsigned)((nthr + 255) / 256), 256, 0, stream>>>(
                    xs, w, zbuf, R, INCH[s]);
            } else {
                gemm64_wmma<<<(unsigned)((R + 31) / 32), 256, 0, stream>>>(
                    xs, w, zbuf, R);
            }
            const int nnz = in_sizes[35 + p];
            long long nthr = (long long)nnz * 32;
            scatter_add<<<(unsigned)((nthr + 255) / 256), 256, 0, stream>>>(
                rows[p], cols[p], vals[p], zbuf,
                acc + (size_t)ROFF[t] * 64, nnz);
        }
        long long n = (long long)FEAT;
        act_kernel<<<(unsigned)((n + 255) / 256), 256, 0, stream>>>(
            acc, cur ? cur : acc, n, (li > 1) ? 1 : 0);
        cur = acc;
    }

    pool_kernel<<<64, 256, 0, stream>>>(cur, 50000, pool);
    head_kernel<<<1, 512, 0, stream>>>(
        pool, gf,
        (const float*)d_in[96],  (const float*)d_in[97],
        (const float*)d_in[98],  (const float*)d_in[99],
        (const float*)d_in[100], (const float*)d_in[101],
        (const float*)d_in[102], (const float*)d_in[103],
        (const float*)d_in[104], (const float*)d_in[105],
        (const float*)d_in[106], (const float*)d_in[107],
        (const float*)d_in[108], (const float*)d_in[109],
        (float*)d_out, 1.0f / 50000.0f);
}
